// ModernNCA_6270652252222
// MI455X (gfx1250) — matrix operand order repaired and checked by reading, not compiled, not run on hardware
//
#include <hip/hip_runtime.h>
#include <hip/hip_bf16.h>

typedef __attribute__((ext_vector_type(16))) _Float16 v16h;
typedef __attribute__((ext_vector_type(8)))  _Float16 v8h;
typedef __attribute__((ext_vector_type(8)))  float    v8f;

#define N_CAND   131072
#define N_Q      512
#define D_ENC    192      // D_OUT = 8*16 + 4*16
#define D_HID    256
#define NSPLIT   64
#define TILES_PER_SPLIT ((N_CAND / NSPLIT) / 16)   // 128

// ---------------- WMMA fragment helpers (layouts per CDNA5 ISA 7.12.2) ----

// 16-bit A-matrix 16x32 (MxK): lane m = lane&15; half h -> K =
// ((h>>3)<<4) | (((lane>>4)&1)<<3) | (h&7).  Per lane this is two
// contiguous 8-half runs -> two 16-byte loads.
__device__ __forceinline__ v16h load_a_frag(const _Float16* base, int ld,
                                            int kbase, int lane) {
  int m   = lane & 15;
  int khi = ((lane >> 4) & 1) << 3;          // +8 for upper half-wave
  const _Float16* p = base + m * ld + kbase + khi;
  v8h lo = *(const v8h*)(p);                 // K = khi + 0..7
  v8h hi = *(const v8h*)(p + 16);            // K = 16 + khi + 0..7
  return __builtin_shufflevector(lo, hi, 0, 1, 2, 3, 4, 5, 6, 7,
                                 8, 9, 10, 11, 12, 13, 14, 15);
}

// B-matrix 32x16 (KxN) where column n is a row of a [rows][ldk] f16 matrix
// (B(k,n) = C[(n0+n)*ldk + kbase + k]) -> two 8-half runs per lane (A's K order).
__device__ __forceinline__ v16h load_b_rows_h(const _Float16* C, size_t ldk,
                                              int n0, size_t kbase, int lane) {
  int n = lane & 15;
  const _Float16* p = C + (size_t)(n0 + n) * ldk + kbase +
                      (((lane >> 4) & 1) << 3);
  v8h lo = *(const v8h*)(p);
  v8h hi = *(const v8h*)(p + 16);
  return __builtin_shufflevector(lo, hi, 0, 1, 2, 3, 4, 5, 6, 7,
                                 8, 9, 10, 11, 12, 13, 14, 15);
}

__device__ __forceinline__ v8f wmma16(v16h a, v16h b, v8f c) {
  return __builtin_amdgcn_wmma_f32_16x16x32_f16(
      /*neg_a=*/false, a, /*neg_b=*/false, b,
      /*c_mod=*/(short)0, c, /*reuse_a=*/false, /*reuse_b=*/false);
}

// ---------------- Kernel 0a: convert+transpose weights to f16 ------------
// W1 [192][256] f32 -> W1T [256][192] f16 ; W2 [256][192] f32 -> W2T [192][256]
__global__ __launch_bounds__(256) void convert_weights_kernel(
    const float* __restrict__ W1, const float* __restrict__ W2,
    _Float16* __restrict__ W1T, _Float16* __restrict__ W2T) {
  int idx = blockIdx.x * 256 + threadIdx.x;
  if (idx < D_ENC * D_HID) {
    int k = idx / D_HID, n = idx % D_HID;
    W1T[(size_t)n * D_ENC + k] = (_Float16)W1[idx];
  }
  if (idx < D_HID * D_ENC) {
    int k = idx / D_ENC, n = idx % D_ENC;
    W2T[(size_t)n * D_HID + k] = (_Float16)W2[idx];
  }
}

// ---------------- Kernel 0b: cand_y -> YT f16 [16][N_CAND] ---------------
// rows 0..9 = labels, row 10 = ones (softmax denominator via same WMMA),
// rows 11..15 = zeros.  Writes coalesced.
__global__ __launch_bounds__(256) void y_transpose_kernel(
    const float* __restrict__ cand_y, _Float16* __restrict__ YT) {
  int idx = blockIdx.x * 256 + threadIdx.x;
  if (idx >= 16 * N_CAND) return;
  int j = idx / N_CAND, cand = idx % N_CAND;
  float v = (j < 10) ? cand_y[(size_t)cand * 10 + j]
                     : (j == 10 ? 1.0f : 0.0f);
  YT[idx] = (_Float16)v;
}

// ---------------- Kernel 1: encode + residual MLP block ------------------
// One wave handles 16 rows. 4 waves / 128 threads per block = 64 rows.
__global__ __launch_bounds__(128) void encode_block_kernel(
    const float* __restrict__ x_num, const int* __restrict__ x_cat,
    const float* __restrict__ W_num, const float* __restrict__ b_num,
    const float* __restrict__ emb,
    const _Float16* __restrict__ W1T, const float* __restrict__ b1,
    const _Float16* __restrict__ W2T, const float* __restrict__ b2,
    _Float16* __restrict__ enc, float* __restrict__ nrm) {
  __shared__ _Float16 zbuf[4][16][D_ENC];   // per-wave private regions
  __shared__ _Float16 hbuf[4][16][D_HID];

  const int tid  = threadIdx.x;
  const int lane = tid & 31;
  const int wave = tid >> 5;
  const int r0   = blockIdx.x * 64 + wave * 16;
  const int hg   = (lane >> 4) & 1;

  _Float16 (*zb)[D_ENC] = zbuf[wave];
  _Float16 (*hb)[D_HID] = hbuf[wave];

  // ---- encode: lane owns row (lane&15), half-group owns half the cols ----
  {
    int m  = lane & 15;
    int gr = r0 + m;
    float xs[8];
#pragma unroll
    for (int j = 0; j < 8; ++j) xs[j] = x_num[gr * 8 + j];
    int cats[4];
#pragma unroll
    for (int j = 0; j < 4; ++j) cats[j] = x_cat[gr * 4 + j];
#pragma unroll
    for (int b8 = 0; b8 < 12; ++b8) {
      int col0 = (hg * 12 + b8) * 8;
      v8h o;
#pragma unroll
      for (int e = 0; e < 8; ++e) {
        int col = col0 + e;
        float v;
        if (col < 128) {                     // numeric: x*W + b
          int j = col >> 4, d = col & 15;
          v = xs[j] * W_num[j * 16 + d] + b_num[j * 16 + d];
        } else {                             // categorical embedding
          int j = (col - 128) >> 4, d = col & 15;
          v = emb[(size_t)(j * 100 + cats[j]) * 16 + d];
        }
        o[e] = (_Float16)v;
      }
      *(v8h*)&zb[m][col0] = o;
    }
  }

  // ---- GEMM1: h = relu(z @ W1 + b1), 16x192 @ 192x256 ----
  v16h za[6];
#pragma unroll
  for (int kk = 0; kk < 6; ++kk)
    za[kk] = load_a_frag(&zb[0][0], D_ENC, kk * 32, lane);

#pragma unroll 1
  for (int nt = 0; nt < 16; ++nt) {
    float bn = b1[nt * 16 + (lane & 15)];
    v8f c;
#pragma unroll
    for (int v = 0; v < 8; ++v) c[v] = bn;
#pragma unroll
    for (int kk = 0; kk < 6; ++kk)
      c = wmma16(za[kk], load_b_rows_h(W1T, D_ENC, nt * 16, kk * 32, lane), c);
    int colc = nt * 16 + (lane & 15);
#pragma unroll
    for (int v = 0; v < 8; ++v) {
      float t = fmaxf(c[v], 0.0f);
      hb[v + 8 * hg][colc] = (_Float16)t;    // C layout -> row-major LDS
    }
  }

  // ---- GEMM2: out = z + h @ W2 + b2, 16x256 @ 256x192 ----
  v16h ha[8];
#pragma unroll
  for (int kk = 0; kk < 8; ++kk)
    ha[kk] = load_a_frag(&hb[0][0], D_HID, kk * 32, lane);

  float norm8[8];
#pragma unroll
  for (int v = 0; v < 8; ++v) norm8[v] = 0.0f;

#pragma unroll 1
  for (int nt = 0; nt < 12; ++nt) {
    float bn = b2[nt * 16 + (lane & 15)];
    v8f c;
#pragma unroll
    for (int v = 0; v < 8; ++v) c[v] = bn;
#pragma unroll
    for (int kk = 0; kk < 8; ++kk)
      c = wmma16(ha[kk], load_b_rows_h(W2T, D_HID, nt * 16, kk * 32, lane), c);
    int colc = nt * 16 + (lane & 15);
#pragma unroll
    for (int v = 0; v < 8; ++v) {
      int row = v + 8 * hg;
      float o = c[v] + (float)zb[row][colc];     // residual
      enc[(size_t)(r0 + row) * D_ENC + colc] = (_Float16)o;
      norm8[v] += o * o;
    }
  }

  // per-row squared norm: reduce across the 16-lane half-group
#pragma unroll
  for (int v = 0; v < 8; ++v) {
#pragma unroll
    for (int off = 8; off >= 1; off >>= 1)
      norm8[v] += __shfl_xor(norm8[v], off, 16);
  }
  if ((lane & 15) == 0) {
#pragma unroll
    for (int v = 0; v < 8; ++v) nrm[r0 + v + 8 * hg] = norm8[v];
  }
}

// ---------------- Kernel 2: fused dist + softmax + probs@Y, all WMMA -----
// logit = -dist <= 0 and dist is O(5) for these 0.05-scaled encodings, so
// exp(logit) is fp32-safe without max-shifting (softmax is shift-invariant).
// Per 16x16 tile: 6 WMMAs for scores, 1 WMMA for P@[Y|1] which yields both
// the numerator (cols 0..9) and denominator (col 10 via the ones-row of YT).
// grid = (32 query-blocks, NSPLIT candidate splits), 128 threads (4 waves).
__global__ __launch_bounds__(128) void nca_partial_kernel(
    const _Float16* __restrict__ xenc, const float* __restrict__ xnrm,
    const _Float16* __restrict__ cenc, const float* __restrict__ cnrm,
    const _Float16* __restrict__ YT,
    float* __restrict__ ps, float* __restrict__ pa) {
  __shared__ _Float16 pbuf[4][16][32];      // per-wave P staging (K padded)
  __shared__ float    red_a[4][16][16];     // per-wave output fragments

  const int qb    = blockIdx.x;
  const int split = blockIdx.y;
  const int tid   = threadIdx.x;
  const int lane  = tid & 31;
  const int wave  = tid >> 5;
  const int hg    = (lane >> 4) & 1;
  const int q0    = qb * 16;

  _Float16 (*pb)[32] = pbuf[wave];

  // zero the P staging buffer once (upper K half stays zero forever)
  {
    v8h z = {};
    for (int idx = lane; idx < (16 * 32) / 8; idx += 32)
      *(v8h*)(&pb[0][0] + idx * 8) = z;
  }

  // query A-fragments, resident for whole kernel
  v16h xa[6];
#pragma unroll
  for (int kk = 0; kk < 6; ++kk)
    xa[kk] = load_a_frag(xenc + (size_t)q0 * D_ENC, D_ENC, kk * 32, lane);

  float xn[8];
#pragma unroll
  for (int v = 0; v < 8; ++v) xn[v] = xnrm[q0 + v + 8 * hg];

  v8f acc;                                  // C-frag: rows=queries, cols=[Y|1]
#pragma unroll
  for (int v = 0; v < 8; ++v) acc[v] = 0.0f;

  const int t0 = split * TILES_PER_SPLIT;
#pragma unroll 1
  for (int t = wave; t < TILES_PER_SPLIT; t += 4) {
    const int cb = (t0 + t) * 16;
    if (t + 4 < TILES_PER_SPLIT)
      __builtin_prefetch(cenc + (size_t)(cb + 64) * D_ENC, 0, 3);

    // scores S = x @ c^T for a 16x16 tile
    v8f S;
#pragma unroll
    for (int v = 0; v < 8; ++v) S[v] = 0.0f;
#pragma unroll
    for (int kk = 0; kk < 6; ++kk)
      S = wmma16(xa[kk], load_b_rows_h(cenc, D_ENC, cb, kk * 32, lane), S);

    const float cn = cnrm[cb + (lane & 15)];

    // p = exp(-sqrt(max(d2,0))), staged to LDS in A layout (row, k=cand)
#pragma unroll
    for (int v = 0; v < 8; ++v) {
      float d2 = xn[v] + cn - 2.0f * S[v];
      float p  = __expf(-__builtin_sqrtf(fmaxf(d2, 0.0f)));  // TEMP == 1
      pb[v + 8 * hg][lane & 15] = (_Float16)p;
    }

    // acc += P @ [Y | ones]  (one WMMA; K=16..31 of P is zero)
    v16h pfrag = load_a_frag(&pb[0][0], 32, 0, lane);
    v16h yfrag = load_b_rows_h(YT, (size_t)N_CAND, 0, (size_t)cb, lane);
    acc = wmma16(pfrag, yfrag, acc);
  }

  // combine the 4 waves' fragments (same 16 rows, disjoint candidates)
#pragma unroll
  for (int v = 0; v < 8; ++v)
    red_a[wave][v + 8 * hg][lane & 15] = acc[v];
  __syncthreads();

  if (tid < 16 * 11) {
    int r = tid / 11, jj = tid % 11;
    float tt = red_a[0][r][jj] + red_a[1][r][jj] +
               red_a[2][r][jj] + red_a[3][r][jj];
    size_t base = ((size_t)qb * NSPLIT + split) * 16 + r;
    if (jj == 10) ps[base] = tt;            // sum of probs (denominator)
    else          pa[base * 10 + jj] = tt;  // numerator
  }
}

// ---------------- Kernel 3: merge splits, normalize, emit (512 x 10) -----
__global__ __launch_bounds__(128) void nca_final_kernel(
    const float* __restrict__ ps, const float* __restrict__ pa,
    float* __restrict__ out) {
  int row = blockIdx.x * 128 + threadIdx.x;
  if (row >= N_Q) return;
  int qb = row >> 4, r = row & 15;
  float S = 0.0f, A[10];
#pragma unroll
  for (int j = 0; j < 10; ++j) A[j] = 0.0f;
  for (int sp = 0; sp < NSPLIT; ++sp) {
    size_t b = ((size_t)qb * NSPLIT + sp) * 16 + r;
    S += ps[b];
#pragma unroll
    for (int j = 0; j < 10; ++j) A[j] += pa[b * 10 + j];
  }
  float inv = 1.0f / S;
#pragma unroll
  for (int j = 0; j < 10; ++j) out[(size_t)row * 10 + j] = A[j] * inv;
}

// ---------------- host launch --------------------------------------------
extern "C" void kernel_launch(void* const* d_in, const int* in_sizes, int n_in,
                              void* d_out, int out_size, void* d_ws,
                              size_t ws_size, hipStream_t stream) {
  const float* x_num  = (const float*)d_in[0];
  const int*   x_cat  = (const int*)d_in[1];
  const float* c_num  = (const float*)d_in[2];
  const int*   c_cat  = (const int*)d_in[3];
  const float* cand_y = (const float*)d_in[4];
  const float* W_num  = (const float*)d_in[5];
  const float* b_num  = (const float*)d_in[6];
  const float* emb    = (const float*)d_in[7];
  const float* W1     = (const float*)d_in[8];
  const float* b1     = (const float*)d_in[9];
  const float* W2     = (const float*)d_in[10];
  const float* b2     = (const float*)d_in[11];

  char* ws = (char*)d_ws;
  _Float16* cENC = (_Float16*)ws; ws += (size_t)N_CAND * D_ENC * 2;  // 50.3 MB
  _Float16* xENC = (_Float16*)ws; ws += (size_t)N_Q    * D_ENC * 2;
  float*    cNRM = (float*)ws;    ws += (size_t)N_CAND * 4;
  float*    xNRM = (float*)ws;    ws += (size_t)N_Q    * 4;
  float*    PS   = (float*)ws;    ws += (size_t)32 * NSPLIT * 16 * 4;
  float*    PA   = (float*)ws;    ws += (size_t)32 * NSPLIT * 16 * 10 * 4;
  _Float16* W1T  = (_Float16*)ws; ws += (size_t)D_ENC * D_HID * 2;
  _Float16* W2T  = (_Float16*)ws; ws += (size_t)D_HID * D_ENC * 2;
  _Float16* YT   = (_Float16*)ws; // 16*N_CAND halfs + 64B tail pad (B-frag
                                  // reads K=16..31 past the last tile)

  convert_weights_kernel<<<(D_ENC * D_HID + 255) / 256, 256, 0, stream>>>(
      W1, W2, W1T, W2T);
  y_transpose_kernel<<<(16 * N_CAND + 255) / 256, 256, 0, stream>>>(
      cand_y, YT);
  encode_block_kernel<<<N_CAND / 64, 128, 0, stream>>>(
      c_num, c_cat, W_num, b_num, emb, W1T, b1, W2T, b2, cENC, cNRM);
  encode_block_kernel<<<N_Q / 64, 128, 0, stream>>>(
      x_num, x_cat, W_num, b_num, emb, W1T, b1, W2T, b2, xENC, xNRM);
  nca_partial_kernel<<<dim3(N_Q / 16, NSPLIT), 128, 0, stream>>>(
      xENC, xNRM, cENC, cNRM, YT, PS, PA);
  nca_final_kernel<<<(N_Q + 127) / 128, 128, 0, stream>>>(
      PS, PA, (float*)d_out);
}
